// HGAT_12043088298235
// MI455X (gfx1250) — matrix operand order repaired and checked by reading, not compiled, not run on hardware
//
#include <hip/hip_runtime.h>
#include <hip/hip_bf16.h>

// ---------------------------------------------------------------------------
// Types for CDNA5 WMMA (wave32): v_wmma_f32_16x16x32_bf16
// ---------------------------------------------------------------------------
typedef __bf16 v16bf __attribute__((ext_vector_type(16)));
typedef float  v8f   __attribute__((ext_vector_type(8)));

union FragBF {
    v16bf v;
    unsigned int d[8];
};

__device__ __forceinline__ unsigned short f2bf(float f) {
    union { float f; unsigned int u; } x;
    x.f = f;
    // round-to-nearest-even fp32 -> bf16
    unsigned int r = x.u + 0x7FFFu + ((x.u >> 16) & 1u);
    return (unsigned short)(r >> 16);
}

#define HC 128
#define NHEAD 4
#define CPH 32
#define TILE_ROWS 128
#define KSTEP 32
#define LDS_STRIDE 40   // ushorts per row: 80 B = 20 dwords; 16-B aligned,
                        // 20*i mod 64 -> 16 distinct banks for 16 lanes

// ---------------------------------------------------------------------------
// Weight pre-pass: Bt[col][k] (bf16, row stride K) = B[k][col] (fp32)
// Tiny (<=128x128); runs once per GEMM, stays L2/WGP$-resident.
// ---------------------------------------------------------------------------
__global__ void hgat_bt_cast(const float* __restrict__ B,
                             unsigned short* __restrict__ Bt,
                             int K, int ncols) {
    int i = blockIdx.x * blockDim.x + threadIdx.x;
    if (i >= K * ncols) return;
    int k = i / ncols;
    int c = i - k * ncols;          // coalesced read of B
    Bt[(size_t)c * K + k] = f2bf(B[i]);
}

// ---------------------------------------------------------------------------
// Tiled GEMM: C[rows, NCT*16] = A[rows, K] * Bt^T[K, NCT*16] (+ bias per col)
// A fp32 -> bf16 staged through LDS (vectorized float4 -> b64 stores).
// B fragments read directly from global bf16 Bt (two b128 loads per frag).
// Block: 256 threads (8 waves); wave w owns rows [16w, 16w+16).
// All NCT B-fragments + A-fragment preloaded, then WMMAs run back-to-back.
// ---------------------------------------------------------------------------
template <int NCT>
__global__ __launch_bounds__(256)
void hgat_gemm_bf16(const float* __restrict__ A,
                    const unsigned short* __restrict__ Bt,
                    float* __restrict__ C, const float* __restrict__ bias,
                    int rows, int K) {
    constexpr int NCOLS = NCT * 16;
    __shared__ unsigned short As[TILE_ROWS * LDS_STRIDE]; // [row][k], padded

    const int tid  = threadIdx.x;
    const int wave = tid >> 5;
    const int lane = tid & 31;
    const int half = lane >> 4;   // 0: lanes 0-15, 1: lanes 16-31
    const int lrow = lane & 15;
    const int rowBase = blockIdx.x * TILE_ROWS;

    v8f zero = {0.f, 0.f, 0.f, 0.f, 0.f, 0.f, 0.f, 0.f};
    v8f acc[NCT];
#pragma unroll
    for (int c = 0; c < NCT; ++c) acc[c] = zero;

    for (int k0 = 0; k0 < K; k0 += KSTEP) {
        // Stage A tile (128 x 32 fp32 -> bf16): 1024 float4 slots, 4/thread.
#pragma unroll
        for (int s = tid; s < TILE_ROWS * KSTEP / 4; s += 256) {
            int r  = s >> 3;            // 8 float4 per 32-wide row
            int kk = (s & 7) << 2;
            int gr = rowBase + r;
            float4 v = make_float4(0.f, 0.f, 0.f, 0.f);
            if (gr < rows) v = *(const float4*)(A + (size_t)gr * K + k0 + kk);
            unsigned int lo = (unsigned int)f2bf(v.x) | ((unsigned int)f2bf(v.y) << 16);
            unsigned int hi = (unsigned int)f2bf(v.z) | ((unsigned int)f2bf(v.w) << 16);
            *(uint2*)&As[r * LDS_STRIDE + kk] = make_uint2(lo, hi);
        }
        __syncthreads();

        // A fragment (16x32 bf16, CDNA5 interleave) = two b128 LDS loads:
        //   dwords j=0..3 at byte (arow*80 + 16*half), j=4..7 at +32.
        FragBF a;
        {
            const int arow = (wave << 4) + lrow;
            const char* base = (const char*)As + arow * (LDS_STRIDE * 2) + (half << 4);
            uint4 a0 = *(const uint4*)base;
            uint4 a1 = *(const uint4*)(base + 32);
            a.d[0] = a0.x; a.d[1] = a0.y; a.d[2] = a0.z; a.d[3] = a0.w;
            a.d[4] = a1.x; a.d[5] = a1.y; a.d[6] = a1.z; a.d[7] = a1.w;
        }

        // Preload all B fragments straight from global bf16 Bt[col][K]:
        // lane col = 16c + lrow, K = k0 + 8*half .. +8 and k0 + 16 + 8*half .. +8.
        FragBF b[NCT];
#pragma unroll
        for (int c = 0; c < NCT; ++c) {
            const int col = (c << 4) + lrow;
            const uint4* bp = (const uint4*)(Bt + (size_t)col * K + k0 + (half << 3));
            uint4 b0 = bp[0];
            uint4 b1 = bp[2];
            b[c].d[0] = b0.x; b[c].d[1] = b0.y; b[c].d[2] = b0.z; b[c].d[3] = b0.w;
            b[c].d[4] = b1.x; b[c].d[5] = b1.y; b[c].d[6] = b1.z; b[c].d[7] = b1.w;
        }

#pragma unroll
        for (int c = 0; c < NCT; ++c) {
            acc[c] = __builtin_amdgcn_wmma_f32_16x16x32_bf16(
                false, a.v, false, b[c].v, (short)0, acc[c], false, false);
        }
        __syncthreads();
    }

    // C fragment layout: VGPR r -> row (r + 8*half), col = lane&15.
    const int grow0 = rowBase + (wave << 4) + (half << 3);
#pragma unroll
    for (int c = 0; c < NCT; ++c) {
        int col  = (c << 4) + lrow;
        float bv = bias ? bias[col] : 0.f;
#pragma unroll
        for (int r = 0; r < 8; ++r) {
            int gr = grow0 + r;
            if (gr < rows) C[(size_t)gr * NCOLS + col] = acc[c][r] + bv;
        }
    }
}

// ---------------------------------------------------------------------------
// a[n,h] = sum_c H[n, h*32+c] * att[h*32+c]
// ---------------------------------------------------------------------------
__global__ void hgat_att_scores(const float* __restrict__ Hmat,
                                const float* __restrict__ att,
                                float* __restrict__ out, int n) {
    int idx = blockIdx.x * blockDim.x + threadIdx.x;
    if (idx >= n * NHEAD) return;
    int node = idx >> 2, h = idx & 3;
    const float* hp = Hmat + (size_t)node * HC + h * CPH;
    const float* ap = att + h * CPH;
    float s = 0.f;
#pragma unroll
    for (int c = 0; c < CPH; ++c) s += hp[c] * ap[c];
    out[idx] = s;
}

// ---------------------------------------------------------------------------
// init accumulators: acc[n*128]=0, mx[n*4]=-inf, dn[n*4]=0
// ---------------------------------------------------------------------------
__global__ void hgat_init(float* __restrict__ acc, float* __restrict__ mx,
                          float* __restrict__ dn, int n) {
    int i = blockIdx.x * blockDim.x + threadIdx.x;
    if (i < n * HC) acc[i] = 0.f;
    if (i < n * NHEAD) { mx[i] = -__builtin_inff(); dn[i] = 0.f; }
}

__device__ __forceinline__ void atomicMaxF32(float* addr, float val) {
    if (val >= 0.f) atomicMax((int*)addr, __float_as_int(val));
    else            atomicMin((unsigned int*)addr, __float_as_uint(val));
}

__device__ __forceinline__ float lrelu(float x, float s) {
    return x > 0.f ? x : x * s;
}

// segment max of leaky_relu(a_s[src]+a_d[dst]) over dst
__global__ void hgat_edge_max(const int* __restrict__ src, const int* __restrict__ dst,
                              const float* __restrict__ as_, const float* __restrict__ ad_,
                              float* __restrict__ mx, int E) {
    int idx = blockIdx.x * blockDim.x + threadIdx.x;
    if (idx >= E * NHEAD) return;
    int e = idx >> 2, h = idx & 3;
    int s = src[e], d = dst[e];
    float a = lrelu(as_[s * NHEAD + h] + ad_[d * NHEAD + h], 0.2f);
    atomicMaxF32(&mx[d * NHEAD + h], a);
}

// ex = exp(alpha - mx[dst]); dn[dst] += ex; store ex per edge
__global__ void hgat_edge_exp(const int* __restrict__ src, const int* __restrict__ dst,
                              const float* __restrict__ as_, const float* __restrict__ ad_,
                              const float* __restrict__ mx, float* __restrict__ exb,
                              float* __restrict__ dn, int E) {
    int idx = blockIdx.x * blockDim.x + threadIdx.x;
    if (idx >= E * NHEAD) return;
    int e = idx >> 2, h = idx & 3;
    int s = src[e], d = dst[e];
    float a = lrelu(as_[s * NHEAD + h] + ad_[d * NHEAD + h], 0.2f);
    float m = mx[d * NHEAD + h];
    if (!(m > -3.0e38f)) m = 0.f;   // isfinite -> 0 (matches reference)
    float ex = __expf(a - m);
    exb[idx] = ex;
    atomicAdd(&dn[d * NHEAD + h], ex);
}

// acc[dst] += Hs[src] * w   (w per head); one wave per edge, 4 ch per lane
__global__ void hgat_edge_agg(const int* __restrict__ src, const int* __restrict__ dst,
                              const float* __restrict__ Hs, const float* __restrict__ exb,
                              const float* __restrict__ dn, float* __restrict__ acc, int E) {
    int idx = blockIdx.x * blockDim.x + threadIdx.x;
    if (idx >= E * 32) return;
    int e = idx >> 5, t = idx & 31;
    int s = src[e], d = dst[e];
    const float* hrow = Hs + (size_t)s * HC;
    float* arow = acc + (size_t)d * HC;
    __builtin_prefetch(hrow, 0, 1);   // global_prefetch_b8
#pragma unroll
    for (int h = 0; h < NHEAD; ++h) {
        float w = exb[e * NHEAD + h] / (dn[d * NHEAD + h] + 1e-16f);
        int ch = h * CPH + t;
        atomicAdd(&arow[ch], hrow[ch] * w);
    }
}

// x = leaky_relu(x + bias, 0.01) in place
__global__ void hgat_finalize(float* __restrict__ x, const float* __restrict__ bias, int n) {
    int i = blockIdx.x * blockDim.x + threadIdx.x;
    if (i >= n * HC) return;
    float v = x[i] + bias[i & (HC - 1)];
    x[i] = lrelu(v, 0.01f);
}

// ---------------------------------------------------------------------------
// Host side
// ---------------------------------------------------------------------------
static inline int cdiv(long long a, long long b) { return (int)((a + b - 1) / b); }

extern "C" void kernel_launch(void* const* d_in, const int* in_sizes, int n_in,
                              void* d_out, int out_size, void* d_ws, size_t ws_size,
                              hipStream_t stream) {
    const float* emb_user  = (const float*)d_in[0];
    const float* emb_movie = (const float*)d_in[1];
    const float* w0_um_src = (const float*)d_in[2];
    const float* w0_um_dst = (const float*)d_in[3];
    const float* a0_um_src = (const float*)d_in[4];
    const float* a0_um_dst = (const float*)d_in[5];
    const float* b0_um     = (const float*)d_in[6];
    const float* w0_mu_src = (const float*)d_in[7];
    const float* w0_mu_dst = (const float*)d_in[8];
    const float* a0_mu_src = (const float*)d_in[9];
    const float* a0_mu_dst = (const float*)d_in[10];
    const float* b0_mu     = (const float*)d_in[11];
    const float* w1_um     = (const float*)d_in[12];
    const float* a1_um_src = (const float*)d_in[13];
    const float* a1_um_dst = (const float*)d_in[14];
    const float* b1_um     = (const float*)d_in[15];
    const float* w1_mu     = (const float*)d_in[16];
    const float* a1_mu_src = (const float*)d_in[17];
    const float* a1_mu_dst = (const float*)d_in[18];
    const float* b1_mu     = (const float*)d_in[19];
    const float* w_lin     = (const float*)d_in[20];
    const float* b_lin     = (const float*)d_in[21];
    const int*   src_um    = (const int*)d_in[24];
    const int*   dst_um    = (const int*)d_in[25];
    const int*   src_mu    = (const int*)d_in[26];
    const int*   dst_mu    = (const int*)d_in[27];

    const int D   = 64;
    const int OUT = 64;
    const int U = in_sizes[0] / D;
    const int M = in_sizes[1] / D;
    const int E = in_sizes[24];

    float* outp   = (float*)d_out;
    float* logits = outp;                               // [M, 64]
    float* m2     = outp + (size_t)M * OUT;             // [M, 128]
    float* u2     = m2   + (size_t)M * HC;              // [U, 128]

    float* ws = (float*)d_ws;
    float* Hu     = ws; ws += (size_t)U * HC;           // src/dst features, user side
    float* Hm     = ws; ws += (size_t)M * HC;           // src/dst features, movie side
    float* u1f    = ws; ws += (size_t)U * HC;           // layer-0 output (users)
    float* m1f    = ws; ws += (size_t)M * HC;           // layer-0 output (movies)
    float* as_buf = ws; ws += (size_t)U * NHEAD;
    float* ad_buf = ws; ws += (size_t)U * NHEAD;
    float* mxb    = ws; ws += (size_t)U * NHEAD;
    float* dnb    = ws; ws += (size_t)U * NHEAD;
    float* exb    = ws; ws += (size_t)E * NHEAD;
    unsigned short* btw = (unsigned short*)ws;          // bf16 B^T scratch (32 KB)
    (void)n_in; (void)out_size; (void)ws_size;

    const int TB = 256;

    auto gemm128 = [&](const float* A, const float* Bw, float* C, const float* bias,
                       int rows, int K) {
        hgat_bt_cast<<<cdiv((long long)K * HC, TB), TB, 0, stream>>>(Bw, btw, K, HC);
        hgat_gemm_bf16<8><<<cdiv(rows, TILE_ROWS), 256, 0, stream>>>(A, btw, C, bias,
                                                                     rows, K);
    };

    auto conv = [&](const float* srcX, int nsrc, const float* Ws,
                    const float* dstX, int ndst, const float* Wd,
                    int Kin,
                    const float* att_s, const float* att_d, const float* bias,
                    const int* esrc, const int* edst,
                    float* Hs, float* Hd, float* accOut) {
        gemm128(srcX, Ws, Hs, nullptr, nsrc, Kin);
        gemm128(dstX, Wd, Hd, nullptr, ndst, Kin);
        hgat_att_scores<<<cdiv((long long)nsrc * NHEAD, TB), TB, 0, stream>>>(
            Hs, att_s, as_buf, nsrc);
        hgat_att_scores<<<cdiv((long long)ndst * NHEAD, TB), TB, 0, stream>>>(
            Hd, att_d, ad_buf, ndst);
        hgat_init<<<cdiv((long long)ndst * HC, TB), TB, 0, stream>>>(
            accOut, mxb, dnb, ndst);
        hgat_edge_max<<<cdiv((long long)E * NHEAD, TB), TB, 0, stream>>>(
            esrc, edst, as_buf, ad_buf, mxb, E);
        hgat_edge_exp<<<cdiv((long long)E * NHEAD, TB), TB, 0, stream>>>(
            esrc, edst, as_buf, ad_buf, mxb, exb, dnb, E);
        hgat_edge_agg<<<cdiv((long long)E * 32, TB), TB, 0, stream>>>(
            esrc, edst, Hs, exb, dnb, accOut, E);
        hgat_finalize<<<cdiv((long long)ndst * HC, TB), TB, 0, stream>>>(
            accOut, bias, ndst);
    };

    // Layer 0: user->movie  (dst = movies) -> m1f
    conv(emb_user, U, w0_um_src, emb_movie, M, w0_um_dst, D,
         a0_um_src, a0_um_dst, b0_um, src_um, dst_um, Hu, Hm, m1f);
    // Layer 0: movie->user  (dst = users) -> u1f
    conv(emb_movie, M, w0_mu_src, emb_user, U, w0_mu_dst, D,
         a0_mu_src, a0_mu_dst, b0_mu, src_mu, dst_mu, Hm, Hu, u1f);
    // Layer 1: user->movie (shared lin w1_um) -> m2 (into d_out)
    conv(u1f, U, w1_um, m1f, M, w1_um, HC,
         a1_um_src, a1_um_dst, b1_um, src_um, dst_um, Hu, Hm, m2);
    // Layer 1: movie->user (shared lin w1_mu) -> u2 (into d_out)
    conv(m1f, M, w1_mu, u1f, U, w1_mu, HC,
         a1_mu_src, a1_mu_dst, b1_mu, src_mu, dst_mu, Hm, Hu, u2);
    // Head: logits = m2 @ w_lin + b_lin (64 cols -> NCT=4)
    hgat_bt_cast<<<cdiv((long long)HC * OUT, TB), TB, 0, stream>>>(w_lin, btw, HC, OUT);
    hgat_gemm_bf16<4><<<cdiv(M, TILE_ROWS), 256, 0, stream>>>(m2, btw, logits,
                                                              b_lin, M, HC);
}
